// NeighborSamplingGCN_53051436040762
// MI455X (gfx1250) — matrix operand order repaired and checked by reading, not compiled, not run on hardware
//
#include <hip/hip_runtime.h>
#include <hip/hip_bf16.h>
#include <cstdint>

// ---- problem dimensions (from setup_inputs) ----
#define N0    292864   // source nodes
#define F_IN  602      // input features
#define KP1   608      // F_IN padded to multiple of 32
#define N1    11264    // layer-1 targets
#define FH    256      // hidden features
#define E1    281600   // layer-1 edges
#define E2    10240    // layer-2 edges
#define NB    1024     // batch (layer-2 targets)
#define F_OUT 41       // output classes
#define OP2   48       // F_OUT padded to multiple of 16
#define KP2   256      // hidden K already multiple of 32

typedef __bf16 bf16;
typedef __attribute__((ext_vector_type(16))) __bf16 v16bf;
typedef __attribute__((ext_vector_type(8)))  __bf16 v8bf;
typedef __attribute__((ext_vector_type(8)))  float  v8f;

__device__ __forceinline__ bf16 signbf(float v) {
  return (v > 0.f) ? (bf16)1.0f : ((v < 0.f) ? (bf16)(-1.0f) : (bf16)0.0f);
}

// ---------------------------------------------------------------------------
// Weight prep: S = sign(W) as bf16, zero-padded to (outFpad x Kpad),
// alpha[o] = mean(|W[o,:]|) (0 for pad rows).
// ---------------------------------------------------------------------------
__global__ void prep_weights(const float* __restrict__ W, int outF, int K, int Kpad,
                             bf16* __restrict__ S, float* __restrict__ alpha) {
  __shared__ float red[256];
  const int o   = blockIdx.x;     // padded output row
  const int tid = threadIdx.x;
  bf16* Srow = S + (size_t)o * Kpad;
  float s = 0.f;
  if (o < outF) {
    const float* Wrow = W + (size_t)o * K;
    for (int k = tid; k < K; k += 256) {
      float w = Wrow[k];
      s += fabsf(w);
      Srow[k] = signbf(w);
    }
    for (int k = K + tid; k < Kpad; k += 256) Srow[k] = (bf16)0.0f;
  } else {
    for (int k = tid; k < Kpad; k += 256) Srow[k] = (bf16)0.0f;
  }
  red[tid] = s;
  __syncthreads();
  for (int off = 128; off > 0; off >>= 1) {
    if (tid < off) red[tid] += red[tid + off];
    __syncthreads();
  }
  if (tid == 0) alpha[o] = (o < outF) ? (red[0] / (float)K) : 0.f;
}

// ---------------------------------------------------------------------------
// Fused: load 16-row strip of X -> row mean -> binarize sign(x-mean) to bf16
// in LDS -> WMMA bf16 GEMM against sign-weight matrices S_l (all rows) and
// S_r (rows < nrows_r).  P_l = Xb @ S_l^T,  P_r = Xb[:nrows_r] @ S_r^T.
// One block = one 16-row M tile, 8 waves split the output column tiles.
// ---------------------------------------------------------------------------
__global__ __launch_bounds__(256) void fused_bin_gemm(
    const float* __restrict__ X, int nrows, int K, int Kpad,
    const bf16* __restrict__ S_l, const bf16* __restrict__ S_r,
    int out_tiles, int nrows_r,
    float* __restrict__ P_l, float* __restrict__ P_r, int ldP)
{
  extern __shared__ __align__(16) char smem[];
  float* xs    = (float*)smem;          // 16 * K      f32 raw strip
  float* smean = xs + 16 * K;           // 16          row means
  bf16*  As    = (bf16*)(smean + 16);   // 16 * Kpad   binarized bf16

  const int tid = threadIdx.x;
  const int m0  = blockIdx.x * 16;

  // cooperative load of the 16 x K strip (single pass over X)
  for (int i = tid; i < 16 * K; i += 256) {
    int r = i / K, c = i - r * K;
    int gr = m0 + r;
    xs[i] = (gr < nrows) ? X[(size_t)gr * K + c] : 0.f;
  }
  __syncthreads();

  const int wv = tid >> 5, lane = tid & 31;
  // row means: wave wv reduces rows 2wv and 2wv+1
  for (int r = wv * 2; r < wv * 2 + 2; ++r) {
    float s = 0.f;
    for (int c = lane; c < K; c += 32) s += xs[r * K + c];
    for (int off = 16; off > 0; off >>= 1) s += __shfl_down(s, off, 32);
    if (lane == 0) smean[r] = s / (float)K;
  }
  __syncthreads();

  // binarize: sign((x-mean)/(std+eps)) == sign(x-mean); zero-fill K..Kpad
  for (int i = tid; i < 16 * Kpad; i += 256) {
    int r = i / Kpad, c = i - r * Kpad;
    As[i] = (c < K) ? signbf(xs[r * K + c] - smean[r]) : (bf16)0.0f;
  }
  __syncthreads();

  const bool do_r        = (m0 < nrows_r);
  const int  total_tiles = do_r ? out_tiles * 2 : out_tiles;
  const int  rA   = lane & 15;
  const int  aoff = (lane < 16) ? 0 : 8;    // A: K chunks {b,b+7} U {b+16,b+23}
  const int  boff = (lane < 16) ? 0 : 8;    // B: K chunks {b,b+7} U {b+16,b+23}

  const bf16* Arow = As + (size_t)rA * Kpad;

  for (int tt = wv; tt < total_tiles; tt += 8) {   // wave-uniform loop
    const bool  is_r = (tt >= out_tiles);
    const int   nt   = is_r ? (tt - out_tiles) : tt;
    const bf16* Sm   = is_r ? S_r : S_l;
    const bf16* Brow = Sm + (size_t)(nt * 16 + rA) * Kpad;

    v8f acc = {};
    for (int kk = 0; kk < Kpad; kk += 32) {
      union { v16bf v; v8bf h[2]; } ua, ub;
      ua.h[0] = *(const v8bf*)(Arow + kk + aoff);        // ds_load_b128
      ua.h[1] = *(const v8bf*)(Arow + kk + aoff + 16);
      ub.h[0] = *(const v8bf*)(Brow + kk + boff);        // global_load_b128
      ub.h[1] = *(const v8bf*)(Brow + kk + boff + 16);
      acc = __builtin_amdgcn_wmma_f32_16x16x32_bf16(
          false, ua.v, false, ub.v, (short)0, acc, false, false);
    }

    float* P   = is_r ? P_r : P_l;
    const int lim   = is_r ? nrows_r : nrows;
    const int col   = nt * 16 + rA;
    const int rbase = m0 + ((lane < 16) ? 0 : 8);  // C/D layout: VGPR v -> M=v(+8)
#pragma unroll
    for (int v = 0; v < 8; ++v) {
      int gr = rbase + v;
      if (gr < lim) P[(size_t)gr * ldP + col] = acc[v];
    }
  }
}

// ---------------------------------------------------------------------------
// Edge aggregation: acc[dst] += P[src] (F features), cnt[dst] += 1.
// One block per edge.
// ---------------------------------------------------------------------------
__global__ void edge_agg(const float* __restrict__ P,
                         const long long* __restrict__ src,
                         const long long* __restrict__ dst,
                         int F, int ldP, int ldA,
                         float* __restrict__ acc, float* __restrict__ cnt)
{
  const int e = blockIdx.x;
  const int s = (int)src[e];
  const int d = (int)dst[e];
  const float* Pr = P + (size_t)s * ldP;
  float* Ar = acc + (size_t)d * ldA;
  for (int f = threadIdx.x; f < F; f += blockDim.x)
    atomicAdd(&Ar[f], Pr[f]);
  if (threadIdx.x == 0) atomicAdd(&cnt[d], 1.0f);
}

// ---------------------------------------------------------------------------
// Layer-1 epilogue: h = relu(al*acc/max(cnt,1) + ar*P_r + b), FH = 256.
// ---------------------------------------------------------------------------
__global__ void finalize1(const float* __restrict__ acc, const float* __restrict__ cnt,
                          const float* __restrict__ Pr, const float* __restrict__ al,
                          const float* __restrict__ ar, const float* __restrict__ b,
                          float* __restrict__ h)
{
  const int i = blockIdx.x * 256 + threadIdx.x;   // i over N1*FH
  const int t = i >> 8, f = i & 255;
  const float c = fmaxf(cnt[t], 1.0f);
  const float v = al[f] * acc[i] / c + ar[f] * Pr[i] + b[f];
  h[i] = fmaxf(v, 0.f);
}

// ---------------------------------------------------------------------------
// Layer-2 epilogue + log_softmax over F_OUT=41 classes. One thread per row.
// ---------------------------------------------------------------------------
__global__ void finalize2_logsoftmax(const float* __restrict__ acc, const float* __restrict__ cnt,
                                     const float* __restrict__ Qr, const float* __restrict__ al,
                                     const float* __restrict__ ar, const float* __restrict__ b,
                                     float* __restrict__ out)
{
  const int t = blockIdx.x * blockDim.x + threadIdx.x;
  if (t >= NB) return;
  const float c = fmaxf(cnt[t], 1.0f);
  float v[F_OUT];
  float m = -1e30f;
#pragma unroll
  for (int f = 0; f < F_OUT; ++f) {
    float x = al[f] * acc[t * OP2 + f] / c + ar[f] * Qr[t * OP2 + f] + b[f];
    v[f] = x;
    m = fmaxf(m, x);
  }
  float se = 0.f;
#pragma unroll
  for (int f = 0; f < F_OUT; ++f) se += expf(v[f] - m);
  const float lse = m + logf(se);
#pragma unroll
  for (int f = 0; f < F_OUT; ++f) out[t * F_OUT + f] = v[f] - lse;
}

// ---------------------------------------------------------------------------
extern "C" void kernel_launch(void* const* d_in, const int* in_sizes, int n_in,
                              void* d_out, int out_size, void* d_ws, size_t ws_size,
                              hipStream_t stream)
{
  (void)in_sizes; (void)n_in; (void)out_size; (void)ws_size;

  const float*     x   = (const float*)d_in[0];
  const long long* ei1 = (const long long*)d_in[1];   // int64 (2, E1)
  const long long* ei2 = (const long long*)d_in[2];   // int64 (2, E2)
  const float*     W1l = (const float*)d_in[5];
  const float*     W1r = (const float*)d_in[6];
  const float*     b1  = (const float*)d_in[7];
  const float*     W2l = (const float*)d_in[8];
  const float*     W2r = (const float*)d_in[9];
  const float*     b2  = (const float*)d_in[10];
  float* out = (float*)d_out;

  // workspace carve-up (256B aligned)
  char*  ws  = (char*)d_ws;
  size_t off = 0;
  auto alloc = [&](size_t bytes) -> char* {
    char* p = ws + off;
    off += (bytes + 255) & ~(size_t)255;
    return p;
  };
  bf16*  S1l  = (bf16*)alloc((size_t)FH * KP1 * sizeof(bf16));
  bf16*  S1r  = (bf16*)alloc((size_t)FH * KP1 * sizeof(bf16));
  float* a1l  = (float*)alloc(FH * sizeof(float));
  float* a1r  = (float*)alloc(FH * sizeof(float));
  bf16*  S2l  = (bf16*)alloc((size_t)OP2 * KP2 * sizeof(bf16));
  bf16*  S2r  = (bf16*)alloc((size_t)OP2 * KP2 * sizeof(bf16));
  float* a2l  = (float*)alloc(OP2 * sizeof(float));
  float* a2r  = (float*)alloc(OP2 * sizeof(float));
  float* P_l  = (float*)alloc((size_t)N0 * FH * sizeof(float));   // 300 MB
  float* P_r  = (float*)alloc((size_t)N1 * FH * sizeof(float));
  float* acc1 = (float*)alloc((size_t)N1 * FH * sizeof(float));
  float* cnt1 = (float*)alloc((size_t)N1 * sizeof(float));
  float* h    = (float*)alloc((size_t)N1 * FH * sizeof(float));
  float* Q_l  = (float*)alloc((size_t)N1 * OP2 * sizeof(float));
  float* Q_r  = (float*)alloc((size_t)NB * OP2 * sizeof(float));
  float* acc2 = (float*)alloc((size_t)NB * OP2 * sizeof(float));
  float* cnt2 = (float*)alloc((size_t)NB * sizeof(float));

  // weight prep
  prep_weights<<<FH,  256, 0, stream>>>(W1l, FH,    F_IN, KP1, S1l, a1l);
  prep_weights<<<FH,  256, 0, stream>>>(W1r, FH,    F_IN, KP1, S1r, a1r);
  prep_weights<<<OP2, 256, 0, stream>>>(W2l, F_OUT, FH,   KP2, S2l, a2l);
  prep_weights<<<OP2, 256, 0, stream>>>(W2r, F_OUT, FH,   KP2, S2r, a2r);

  // zero accumulators (graph-capturable)
  hipMemsetAsync(acc1, 0, (size_t)N1 * FH  * sizeof(float), stream);
  hipMemsetAsync(cnt1, 0, (size_t)N1       * sizeof(float), stream);
  hipMemsetAsync(acc2, 0, (size_t)NB * OP2 * sizeof(float), stream);
  hipMemsetAsync(cnt2, 0, (size_t)NB       * sizeof(float), stream);

  // layer 1: fused norm->bin->WMMA GEMM (single pass over x)
  {
    size_t shm = (size_t)16 * F_IN * 4 + 16 * 4 + (size_t)16 * KP1 * 2;  // 58,048 B
    fused_bin_gemm<<<N0 / 16, 256, shm, stream>>>(
        x, N0, F_IN, KP1, S1l, S1r, FH / 16, N1, P_l, P_r, FH);
  }
  edge_agg<<<E1, 256, 0, stream>>>(P_l, ei1, ei1 + E1, FH, FH, FH, acc1, cnt1);
  finalize1<<<(N1 * FH) / 256, 256, 0, stream>>>(acc1, cnt1, P_r, a1l, a1r, b1, h);

  // layer 2
  {
    size_t shm = (size_t)16 * KP2 * 4 + 16 * 4 + (size_t)16 * KP2 * 2;   // 24,640 B
    fused_bin_gemm<<<N1 / 16, 256, shm, stream>>>(
        h, N1, KP2, KP2, S2l, S2r, OP2 / 16, NB, Q_l, Q_r, OP2);
  }
  edge_agg<<<E2, 64, 0, stream>>>(Q_l, ei2, ei2 + E2, F_OUT, OP2, OP2, acc2, cnt2);
  finalize2_logsoftmax<<<(NB + 255) / 256, 256, 0, stream>>>(
      acc2, cnt2, Q_r, a2l, a2r, b2, out);
}
